// ml_mlp_59777354826018
// MI455X (gfx1250) — matrix operand order repaired and checked by reading, not compiled, not run on hardware
//
#include <hip/hip_runtime.h>
#include <math.h>

// ---------------------------------------------------------------------------
// Problem constants (match reference)
// ---------------------------------------------------------------------------
#define BATCH   1024
#define IN_DIMS 784
#define WIDTH   128
#define DEPTH   10
#define NCLS    10

// Sigma kernel tiling
#define PITCH   136                 // halves per LDS matrix row (bank-conflict pad)
#define BUF     (WIDTH * PITCH)     // halves per 128x128 LDS matrix buffer
#define NITER   16                  // Newton-Schulz iterations

typedef _Float16 v8h  __attribute__((ext_vector_type(8)));
typedef _Float16 v16h __attribute__((ext_vector_type(16)));
typedef float    v8f  __attribute__((ext_vector_type(8)));

// ---------------------------------------------------------------------------
// WMMA fragment helpers (ISA 7.12.2 wave32 layouts, 16x16x32 f16)
// ---------------------------------------------------------------------------
// A-operand: 16x32 tile, rows of a row-major buffer.
__device__ __forceinline__ v16h frag_a(const _Float16* m, int row0, int k0, int lane) {
  const _Float16* p = m + (size_t)(row0 + (lane & 15)) * PITCH + k0 + ((lane & 16) ? 8 : 0);
  v8h lo = *(const v8h*)(p);
  v8h hi = *(const v8h*)(p + 16);
  return __builtin_shufflevector(lo, hi, 0,1,2,3,4,5,6,7,8,9,10,11,12,13,14,15);
}

// B-operand: 32x16 tile of Q, fetched from a buffer holding Q^T row-major
// (for symmetric Q the buffer is Q itself).
__device__ __forceinline__ v16h frag_bt(const _Float16* m, int col0, int k0, int lane) {
  const _Float16* p = m + (size_t)(col0 + (lane & 15)) * PITCH + k0 + ((lane & 16) ? 8 : 0);
  v8h lo = *(const v8h*)(p);
  v8h hi = *(const v8h*)(p + 16);
  return __builtin_shufflevector(lo, hi, 0,1,2,3,4,5,6,7,8,9,10,11,12,13,14,15);
}

__device__ __forceinline__ v8f wmma16(v16h a, v16h b, v8f c) {
  return __builtin_amdgcn_wmma_f32_16x16x32_f16(false, a, false, b, (short)0, c,
                                                false, false);
}

__device__ __forceinline__ float wave_red(float v) {
#pragma unroll
  for (int m = 16; m >= 1; m >>= 1) v += __shfl_xor(v, m, 32);
  return v;
}

// ---------------------------------------------------------------------------
// Sigma kernel: one workgroup (256 threads = 8 wave32) per batch element.
//   J^T built in LDS (f16);  A = J^T J via WMMA (exact);
//   sum(sigma^2) = trace(A);  sum(sigma) = ||A||_F^(1/2) * trace(NS-sqrt(A/||A||_F))
// All NS iterates are symmetric, so f16 results are stored TRANSPOSED:
// a lane's 8 tile outputs (same column, consecutive rows) become one
// contiguous ds_store_b128 instead of 8 ds_store_b16.
// ---------------------------------------------------------------------------
__global__ __launch_bounds__(256) void sigma_kernel(const float* __restrict__ Wd,
                                                    const float* __restrict__ y,
                                                    float* __restrict__ gacc) {
  extern __shared__ _Float16 lds[];
  _Float16* b0 = lds;                 // Y (cur)
  _Float16* b1 = lds + 1 * BUF;       // Z (cur)
  _Float16* bT = lds + 2 * BUF;       // T / initial J^T
  _Float16* b3 = lds + 3 * BUF;       // Y (next)
  _Float16* b4 = lds + 4 * BUF;       // Z (next)
  float*    red = (float*)(lds + 5 * BUF);   // [0]=frob2 [1]=trA [2]=trY
  float*    msk = red + 8;                   // 128 floats

  const int tid  = threadIdx.x;
  const int lane = tid & 31;
  const int wave = tid >> 5;
  const int b    = blockIdx.x;

  if (tid < WIDTH) msk[tid] = (y[(size_t)b * WIDTH + tid] >= 0.f) ? 1.f : 0.f;
  if (tid < 4)     red[tid] = 0.f;
  __syncthreads();

  // Build J^T in LDS (f16):  Jt[i][j] = W[j][i]*m[i] + (i==j)
  for (int idx = tid; idx < WIDTH * WIDTH; idx += 256) {
    int j = idx >> 7, i = idx & 127;               // coalesced over W row j
    float v = Wd[j * WIDTH + i] * msk[i] + ((i == j) ? 1.f : 0.f);
    bT[(size_t)i * PITCH + j] = (_Float16)v;
  }
  __syncthreads();

  const int ti    = wave;           // this wave owns output tile-row ti
  const int row0  = ti * 16;
  const int n_    = lane & 15;
  const int off   = (lane & 16) ? 8 : 0;
  const int rbase = row0 + off;     // first of the 8 rows this lane holds

  // ---- Gram: A = (J^T) * (J^T)^T, keep tiles in registers until norm known --
  v8f accs[8];
  float frob = 0.f, trA = 0.f;
  {
    v16h aj[4];
#pragma unroll
    for (int k = 0; k < 4; ++k) aj[k] = frag_a(bT, row0, k * 32, lane);
#pragma unroll
    for (int tj = 0; tj < 8; ++tj) {
      v8f acc = {0.f, 0.f, 0.f, 0.f, 0.f, 0.f, 0.f, 0.f};
#pragma unroll
      for (int k = 0; k < 4; ++k)
        acc = wmma16(aj[k], frag_bt(bT, tj * 16, k * 32, lane), acc);
      accs[tj] = acc;
#pragma unroll
      for (int r = 0; r < 8; ++r) frob += acc[r] * acc[r];
      if (tj == ti) {   // diagonal elements: lanes 0-7 (r=lane), lanes 24-31 (r=lane-24)
        int r = (lane < 16) ? lane : lane - 24;
        if (r >= 0 && r < 8) trA += acc[r];
      }
    }
  }
  frob = wave_red(frob);
  trA  = wave_red(trA);
  if (lane == 0) {
    atomicAdd(&red[0], frob);
    atomicAdd(&red[1], trA);
  }
  __syncthreads();
  const float c     = sqrtf(red[0]);     // ||A||_F  (>0: diag(A) ~ 1)
  const float inv_c = 1.f / c;

  // Y0 = A/c (f16, stored transposed => one b128 store per tile), Z0 = I
#pragma unroll
  for (int tj = 0; tj < 8; ++tj) {
    int col = tj * 16 + n_;
    v8h pk;
#pragma unroll
    for (int r = 0; r < 8; ++r) pk[r] = (_Float16)(accs[tj][r] * inv_c);
    *(v8h*)(b0 + (size_t)col * PITCH + rbase) = pk;
  }
  for (int idx = tid; idx < WIDTH * (WIDTH / 8); idx += 256) {
    int i2 = idx >> 4, j8 = (idx & 15) * 8;
    v8h pk;
#pragma unroll
    for (int e = 0; e < 8; ++e) pk[e] = (_Float16)((i2 == j8 + e) ? 1.f : 0.f);
    *(v8h*)(b1 + (size_t)i2 * PITCH + j8) = pk;
  }
  __syncthreads();

  // ---- Newton-Schulz: T = 3I - Z*Y ; Y' = 0.5*Y*T ; Z' = 0.5*Z*T (commuting) --
  _Float16 *Yc = b0, *Zc = b1, *Yn = b3, *Zn = b4;
  float trY = 0.f;
  for (int it = 0; it < NITER; ++it) {
    {  // T = 3I - Zc*Yc   (stored transposed, symmetric)
      v16h az[4];
#pragma unroll
      for (int k = 0; k < 4; ++k) az[k] = frag_a(Zc, row0, k * 32, lane);
#pragma unroll
      for (int tj = 0; tj < 8; ++tj) {
        v8f acc = {0.f, 0.f, 0.f, 0.f, 0.f, 0.f, 0.f, 0.f};
#pragma unroll
        for (int k = 0; k < 4; ++k)
          acc = wmma16(az[k], frag_bt(Yc, tj * 16, k * 32, lane), acc);
        int col = tj * 16 + n_;
        v8h pk;
#pragma unroll
        for (int r = 0; r < 8; ++r)
          pk[r] = (_Float16)(((rbase + r == col) ? 3.f : 0.f) - acc[r]);
        *(v8h*)(bT + (size_t)col * PITCH + rbase) = pk;
      }
    }
    __syncthreads();
    {  // Yn = 0.5*Yc*T ; Zn = 0.5*Zc*T   (stored transposed, symmetric)
      const bool last = (it == NITER - 1);
      v16h ay[4], az[4];
#pragma unroll
      for (int k = 0; k < 4; ++k) {
        ay[k] = frag_a(Yc, row0, k * 32, lane);
        az[k] = frag_a(Zc, row0, k * 32, lane);
      }
#pragma unroll
      for (int tj = 0; tj < 8; ++tj) {
        v8f ay8 = {0.f, 0.f, 0.f, 0.f, 0.f, 0.f, 0.f, 0.f};
        v8f az8 = {0.f, 0.f, 0.f, 0.f, 0.f, 0.f, 0.f, 0.f};
#pragma unroll
        for (int k = 0; k < 4; ++k) {
          v16h bf = frag_bt(bT, tj * 16, k * 32, lane);
          ay8 = wmma16(ay[k], bf, ay8);
          az8 = wmma16(az[k], bf, az8);
        }
        int col = tj * 16 + n_;
        v8h pky, pkz;
#pragma unroll
        for (int r = 0; r < 8; ++r) {
          float vy = 0.5f * ay8[r];
          pky[r] = (_Float16)vy;
          pkz[r] = (_Float16)(0.5f * az8[r]);
          if (last && (rbase + r == col)) trY += vy;
        }
        *(v8h*)(Yn + (size_t)col * PITCH + rbase) = pky;
        *(v8h*)(Zn + (size_t)col * PITCH + rbase) = pkz;
      }
    }
    __syncthreads();
    _Float16* t;
    t = Yc; Yc = Yn; Yn = t;
    t = Zc; Zc = Zn; Zn = t;
  }

  trY = wave_red(trY);
  if (lane == 0) atomicAdd(&red[2], trY);
  __syncthreads();
  if (tid == 0) {
    atomicAdd(&gacc[0], sqrtf(c) * red[2]);  // sum sigma   (trace A^(1/2))
    atomicAdd(&gacc[1], red[1]);             // sum sigma^2 (trace A, exact)
  }
}

// ---------------------------------------------------------------------------
// Forward-pass kernels (f32 SIMT; tiny workload, precision-critical)
// ---------------------------------------------------------------------------
__global__ void zero_acc(float* g) {
  if (threadIdx.x < 2) g[threadIdx.x] = 0.f;
}

__global__ __launch_bounds__(256) void first_layer(const float* __restrict__ x,
                                                   const float* __restrict__ W,
                                                   const float* __restrict__ bias,
                                                   float* __restrict__ yout) {
  int g = blockIdx.x * 256 + threadIdx.x;   // 1024*128
  int o = g & 127, b = g >> 7;
  const float4* xr = (const float4*)(x + (size_t)b * IN_DIMS);
  const float4* wr = (const float4*)(W + (size_t)o * IN_DIMS);
  float s = bias[o];
  for (int k = 0; k < IN_DIMS / 4; ++k) {
    float4 xv = xr[k], wv = wr[k];
    s += xv.x * wv.x + xv.y * wv.y + xv.z * wv.z + xv.w * wv.w;
  }
  yout[g] = fmaxf(s, 0.f);
}

__global__ __launch_bounds__(256) void block_layer(const float* __restrict__ yin,
                                                   const float* __restrict__ W,
                                                   const float* __restrict__ bias,
                                                   float* __restrict__ yout) {
  int g = blockIdx.x * 256 + threadIdx.x;   // 1024*128
  int o = g & 127, b = g >> 7;
  const float*  yr  = yin + (size_t)b * WIDTH;
  const float4* yr4 = (const float4*)yr;
  const float4* wr4 = (const float4*)(W + (size_t)o * WIDTH);
  float s = bias[o];
#pragma unroll 8
  for (int k = 0; k < WIDTH / 4; ++k) {
    float4 yv = yr4[k], wv = wr4[k];
    s += yv.x * wv.x + yv.y * wv.y + yv.z * wv.z + yv.w * wv.w;
  }
  yout[g] = fmaxf(s, 0.f) + yr[o];
}

__global__ __launch_bounds__(256) void last_layer(const float* __restrict__ yin,
                                                  const float* __restrict__ W,
                                                  const float* __restrict__ bias,
                                                  float* __restrict__ out) {
  int g = blockIdx.x * 256 + threadIdx.x;   // 1024*10
  if (g >= BATCH * NCLS) return;
  int cidx = g % NCLS, b = g / NCLS;
  const float* yr = yin + (size_t)b * WIDTH;
  const float* wr = W + (size_t)cidx * WIDTH;
  float s = bias[cidx];
#pragma unroll 8
  for (int k = 0; k < WIDTH; ++k) s += yr[k] * wr[k];
  out[g] = s;
}

__global__ void finalize(const float* __restrict__ gacc, float* __restrict__ out2) {
  if (threadIdx.x == 0) {
    const float N = (float)DEPTH * (float)BATCH * (float)WIDTH;
    float mean = gacc[0] / N;
    float var  = gacc[1] / N - mean * mean;
    out2[0] = mean;
    out2[1] = sqrtf(fmaxf(var, 0.f));
  }
}

// ---------------------------------------------------------------------------
// Launch
// ---------------------------------------------------------------------------
extern "C" void kernel_launch(void* const* d_in, const int* in_sizes, int n_in,
                              void* d_out, int out_size, void* d_ws, size_t ws_size,
                              hipStream_t stream) {
  (void)in_sizes; (void)n_in; (void)out_size; (void)ws_size;
  const float* x       = (const float*)d_in[0];
  const float* W_first = (const float*)d_in[1];
  const float* b_first = (const float*)d_in[2];
  const float* W_blk   = (const float*)d_in[3];
  const float* b_blk   = (const float*)d_in[4];
  const float* W_last  = (const float*)d_in[5];
  const float* b_last  = (const float*)d_in[6];
  float* out = (float*)d_out;

  float* gacc = (float*)d_ws;                       // 2 accumulators
  float* y0   = (float*)((char*)d_ws + 256);        // [1024,128]
  float* y1   = y0 + (size_t)BATCH * WIDTH;         // [1024,128]

  const size_t lds_bytes = (size_t)5 * BUF * sizeof(_Float16) + (8 + WIDTH) * sizeof(float);
  (void)hipFuncSetAttribute((const void*)sigma_kernel,
                            hipFuncAttributeMaxDynamicSharedMemorySize,
                            (int)lds_bytes);

  zero_acc<<<1, 32, 0, stream>>>(gacc);
  first_layer<<<(BATCH * WIDTH) / 256, 256, 0, stream>>>(x, W_first, b_first, y0);

  float* cur = y0;
  float* nxt = y1;
  for (int d = 0; d < DEPTH; ++d) {
    const float* Wd = W_blk + (size_t)d * WIDTH * WIDTH;
    sigma_kernel<<<BATCH, 256, lds_bytes, stream>>>(Wd, cur, gacc);
    block_layer<<<(BATCH * WIDTH) / 256, 256, 0, stream>>>(cur, Wd, b_blk + (size_t)d * WIDTH, nxt);
    float* t = cur; cur = nxt; nxt = t;
  }

  last_layer<<<(BATCH * NCLS + 255) / 256, 256, 0, stream>>>(cur, W_last, b_last, out);
  finalize<<<1, 32, 0, stream>>>(gacc, out + BATCH * NCLS);
}
